// Net_1632087572624
// MI455X (gfx1250) — matrix operand order repaired and checked by reading, not compiled, not run on hardware
//
#include <hip/hip_runtime.h>
#include <cstdint>

// ---------------------------------------------------------------------------
// MNIST-superpixel SplineConv net on gfx1250.
//  - scatter/pool stages: VALU + float/uint atomics
//  - all dense transforms (spline einsum folded with root/bias, FC layers):
//    v_wmma_f32_16x16x32_f16; each wave owns a 16xN strip (N/16 WMMAs share
//    one A fragment); B tile double-buffered through LDS (1 barrier/K-step).
// ---------------------------------------------------------------------------

typedef __attribute__((ext_vector_type(16))) _Float16 v16h;
typedef __attribute__((ext_vector_type(8)))  _Float16 v8h;
typedef __attribute__((ext_vector_type(8)))  float    v8f;

#define TPB 256

// ---- monotone order-preserving float<->uint encoding (for atomic max) -----
__device__ __forceinline__ unsigned enc_ord(float f) {
  unsigned u = __float_as_uint(f);
  return (u & 0x80000000u) ? ~u : (u | 0x80000000u);
}
__device__ __forceinline__ float dec_ord(unsigned u) {
  unsigned b = (u & 0x80000000u) ? (u & 0x7FFFFFFFu) : ~u;
  return __uint_as_float(b);
}
__device__ __forceinline__ float load_amax(const unsigned* p) {
  unsigned u = *p;
  float v = (u & 0x80000000u) ? __uint_as_float(u & 0x7FFFFFFFu) : 0.0f;
  return fmaxf(v, 1e-12f);
}

// ---- B-spline (dim=2, K=5, degree=1) basis: 4 (weight, kernel-idx) pairs --
__device__ __forceinline__ void spline_basis(float p0, float p1, float* w, int* kk) {
  float v0 = fminf(fmaxf(p0, 0.f), 1.f) * 4.f;
  float v1 = fminf(fmaxf(p1, 0.f), 1.f) * 4.f;
  float b0 = fminf(floorf(v0), 3.f);
  float b1 = fminf(floorf(v1), 3.f);
  float f0 = v0 - b0, f1 = v1 - b1;
  int i0 = (int)b0, i1 = (int)b1;
  w[0] = (1.f - f0) * (1.f - f1); kk[0] = i0     + 5 * i1;
  w[1] = f0         * (1.f - f1); kk[1] = i0 + 1 + 5 * i1;
  w[2] = (1.f - f0) * f1;         kk[2] = i0     + 5 * (i1 + 1);
  w[3] = f0         * f1;         kk[3] = i0 + 1 + 5 * (i1 + 1);
}

// ---------------------------------------------------------------------------
__global__ void k_fill_u32(unsigned* p, unsigned v, long long n) {
  long long i = (long long)blockIdx.x * blockDim.x + threadIdx.x;
  if (i < n) p[i] = v;
}

__global__ void k_amax0(const float* pos, const int* src, const int* dst,
                        long long E, unsigned* out) {
  __shared__ float red[TPB];
  long long e = (long long)blockIdx.x * blockDim.x + threadIdx.x;
  float m = 0.f;
  if (e < E) {
    int s = src[e], d = dst[e];
    float c0 = pos[2 * s] - pos[2 * d];
    float c1 = pos[2 * s + 1] - pos[2 * d + 1];
    m = fmaxf(fabsf(c0), fabsf(c1));
  }
  red[threadIdx.x] = m;
  __syncthreads();
  for (int o = TPB / 2; o > 0; o >>= 1) {
    if ((int)threadIdx.x < o) red[threadIdx.x] = fmaxf(red[threadIdx.x], red[threadIdx.x + o]);
    __syncthreads();
  }
  if (threadIdx.x == 0) atomicMax(out, enc_ord(red[0]));
}

// layer-1 scatter: Fin==1, one thread per edge
__global__ void k_scatter1(const float* x, const float* pos, const int* src, const int* dst,
                           long long E, const unsigned* amaxu, float* acc, float* deg) {
  long long e = (long long)blockIdx.x * blockDim.x + threadIdx.x;
  if (e >= E) return;
  int s = src[e], d = dst[e];
  float am = load_amax(amaxu);
  float p0 = (pos[2 * s] - pos[2 * d]) / (2.f * am) + 0.5f;
  float p1 = (pos[2 * s + 1] - pos[2 * d + 1]) / (2.f * am) + 0.5f;
  float w[4]; int kk[4];
  spline_basis(p0, p1, w, kk);
  float xj = x[s];
  #pragma unroll
  for (int t = 0; t < 4; ++t)
    atomicAdd(&acc[(size_t)d * 25 + kk[t]], w[t] * xj);
  atomicAdd(&deg[d], 1.f);
}

// generic scatter for layers 2/3: one thread per (edge, feature)
__global__ void k_scatter(const float* px, const float* ppos, const int* es, const int* ed,
                          const float* kf, long long E, int F, int fsh,
                          const unsigned* amaxu, float* acc) {
  long long idx = (long long)blockIdx.x * blockDim.x + threadIdx.x;
  long long e = idx >> fsh;
  int f = (int)(idx & (F - 1));
  if (e >= E) return;
  if (kf[e] <= 0.f) return;
  int ns = es[e], nd = ed[e];
  float am = load_amax(amaxu);
  float p0 = (ppos[2 * ns] - ppos[2 * nd]) / (2.f * am) + 0.5f;
  float p1 = (ppos[2 * ns + 1] - ppos[2 * nd + 1]) / (2.f * am) + 0.5f;
  float w[4]; int kk[4];
  spline_basis(p0, p1, w, kk);
  float xj = px[(size_t)ns * F + f];
  #pragma unroll
  for (int t = 0; t < 4; ++t)
    atomicAdd(&acc[(size_t)nd * (25 * F) + (size_t)kk[t] * F + f], w[t] * xj);
}

// voxel-grid pool assign: one thread per (node, feature)
__global__ void k_pool_assign(const float* x, const float* pos, const float* sval_in,
                              long long Min, int F, int node_div, float inv_size, int G,
                              unsigned* px_enc, float* cnt, float* ppos_sum, int* cl_out) {
  long long idx = (long long)blockIdx.x * blockDim.x + threadIdx.x;
  long long n = idx / F;
  int f = (int)(idx % F);
  if (n >= Min) return;
  float p0 = pos[2 * n], p1 = pos[2 * n + 1];
  int c0 = min(max((int)floorf(p0 * inv_size), 0), G - 1);
  int c1 = min(max((int)floorf(p1 * inv_size), 0), G - 1);
  int cl = (int)(n / node_div) * (G * G) + c1 * G + c0;
  if (f == 0 && cl_out) cl_out[n] = cl;
  float valid = sval_in ? sval_in[n] : 1.f;
  if (valid > 0.f) {
    atomicMax(&px_enc[(size_t)cl * F + f], enc_ord(x[(size_t)n * F + f]));
    if (f == 0) {
      atomicAdd(&cnt[cl], 1.f);
      atomicAdd(&ppos_sum[2 * cl], p0);
      atomicAdd(&ppos_sum[2 * cl + 1], p1);
    }
  }
}

// finalize pool: decode max (in place), mean positions, valid mask
__global__ void k_pool_final(unsigned* px_enc, const float* cnt, long long S, int F,
                             float* px, float* ppos, float* sval) {
  long long idx = (long long)blockIdx.x * blockDim.x + threadIdx.x;
  long long n = idx / F;
  int f = (int)(idx % F);
  if (n >= S) return;
  float c = cnt[n];
  bool valid = c > 0.f;
  px[(size_t)n * F + f] = valid ? dec_ord(px_enc[(size_t)n * F + f]) : 0.f;
  if (f == 0) {
    if (sval) sval[n] = valid ? 1.f : 0.f;
    if (ppos) {
      float inv = 1.f / fmaxf(c, 1.f);
      ppos[2 * n] *= inv;
      ppos[2 * n + 1] *= inv;
    }
  }
}

// edge coalesce pass 1: elect one representative per (nsrc,ndst) cluster pair
__global__ void k_edge_remap(const int* es, const int* ed, const float* em, const int* cl,
                             long long E, int G2, unsigned* table) {
  long long e = (long long)blockIdx.x * blockDim.x + threadIdx.x;
  if (e >= E) return;
  float m = em ? em[e] : 1.f;
  int ns = cl[es[e]], nd = cl[ed[e]];
  if (m > 0.f && ns != nd) {
    unsigned key = (unsigned)ns * (unsigned)G2 + (unsigned)(nd % G2);
    atomicMin(&table[key], (unsigned)e);
  }
}

// edge coalesce pass 2: keep mask, deg of kept edges, amax of kept carts
__global__ void k_edge_keep(const int* es, const int* ed, const float* em, const int* cl,
                            long long E, int G2, const unsigned* table, const float* ppos,
                            int* es_out, int* ed_out, float* kf, float* deg, unsigned* amax_out) {
  __shared__ float red[TPB];
  long long e = (long long)blockIdx.x * blockDim.x + threadIdx.x;
  float lmax = 0.f;
  if (e < E) {
    int ns = cl[es[e]], nd = cl[ed[e]];
    es_out[e] = ns;
    ed_out[e] = nd;
    float m = em ? em[e] : 1.f;
    bool keep = false;
    if (m > 0.f && ns != nd) {
      unsigned key = (unsigned)ns * (unsigned)G2 + (unsigned)(nd % G2);
      keep = (table[key] == (unsigned)e);
    }
    kf[e] = keep ? 1.f : 0.f;
    if (keep) {
      atomicAdd(&deg[nd], 1.f);
      float c0 = ppos[2 * ns] - ppos[2 * nd];
      float c1 = ppos[2 * ns + 1] - ppos[2 * nd + 1];
      lmax = fmaxf(fabsf(c0), fabsf(c1));
    }
  }
  red[threadIdx.x] = lmax;
  __syncthreads();
  for (int o = TPB / 2; o > 0; o >>= 1) {
    if ((int)threadIdx.x < o) red[threadIdx.x] = fmaxf(red[threadIdx.x], red[threadIdx.x + o]);
    __syncthreads();
  }
  if (threadIdx.x == 0) atomicMax(amax_out, enc_ord(red[0]));
}

// pack A = [acc/deg | xroot | 0-pad] -> f16, row-major [M, Kpad]
__global__ void k_packA(const float* acc, const float* deg, const float* xroot,
                        long long M, int Ka, int Fr, int Kpad, _Float16* A) {
  long long idx = (long long)blockIdx.x * blockDim.x + threadIdx.x;
  long long n = idx / Kpad;
  int j = (int)(idx % Kpad);
  if (n >= M) return;
  float v = 0.f;
  if (j < Ka) v = acc[(size_t)n * Ka + j] / fmaxf(deg[n], 1.f);
  else if (j < Ka + Fr) v = xroot[(size_t)n * Fr + (j - Ka)];
  A[(size_t)n * Kpad + j] = (_Float16)v;
}

// pack B = [W ; R ; 0-pad] (optionally W transposed) -> f16 [Kpad, Nout]
__global__ void k_packB(const float* W, const float* R, int Kw, int Kr, int Kpad,
                        int Nout, int transW, int ldw, int Nw, _Float16* Bf) {
  long long idx = (long long)blockIdx.x * blockDim.x + threadIdx.x;
  int j = (int)(idx / Nout);
  int nn = (int)(idx % Nout);
  if (j >= Kpad) return;
  float v = 0.f;
  if (j < Kw) {
    if (transW) { if (nn < Nw) v = W[(size_t)nn * ldw + j]; }
    else v = W[(size_t)j * Nout + nn];
  } else if (j < Kw + Kr && R) {
    v = R[(size_t)(j - Kw) * Nout + nn];
  }
  Bf[(size_t)j * Nout + nn] = (_Float16)v;
}

__global__ void k_convert_f16(const float* s, _Float16* d, long long n) {
  long long i = (long long)blockIdx.x * blockDim.x + threadIdx.x;
  if (i < n) d[i] = (_Float16)s[i];
}

__global__ void k_padvec(const float* s, int nvalid, int ntot, float* d) {
  int i = blockIdx.x * blockDim.x + threadIdx.x;
  if (i < ntot) d[i] = (i < nvalid) ? s[i] : 0.f;
}

// ---------------------------------------------------------------------------
// WMMA GEMM: C[M, NT*16] = elu?(A[M,Kpad] x B[Kpad,NT*16] + bias).
// 4 waves/block, each wave owns a 16-row x NT*16-col strip: one A fragment
// feeds NT v_wmma_f32_16x16x32_f16. B tile (32 x NT*16) is double-buffered
// through LDS with a single barrier per K-step.
// ---------------------------------------------------------------------------
template <int NT>
__global__ __launch_bounds__(128) void k_gemm_t(const _Float16* __restrict__ A,
                                                const _Float16* __restrict__ Bm,
                                                const float* __restrict__ bias,
                                                float* __restrict__ C,
                                                int M, int Kpad, int act) {
  constexpr int Nn = NT * 16;
  constexpr int CH = Nn / 4;  // halves staged per thread per K-step
  __shared__ __align__(16) _Float16 ldsBt[2][Nn * 32];  // [buf][col][k]
  int tid = threadIdx.x;
  int wave = tid >> 5, lid = tid & 31;
  int col16 = lid & 15, khalf = lid >> 4;
  int m0 = (blockIdx.x * 4 + wave) * 16;
  int rowm = m0 + col16;
  if (rowm >= M) rowm = M - 1;
  const _Float16* arow = A + (size_t)rowm * Kpad;
  int lk = tid >> 2;  // 0..31: B-tile row
  int lq = tid & 3;   // 0..3 : column quarter
  v8f acc[NT] = {};
  _Float16 breg[CH];
  {  // preload K-tile 0 of B into registers
    const _Float16* brow = Bm + (size_t)lk * Nn + lq * CH;
    #pragma unroll
    for (int i = 0; i < CH; ++i) breg[i] = brow[i];
  }
  int p = 0;
  for (int kb = 0; kb < Kpad; kb += 32) {
    // commit staged tile to LDS (transposed: [col][k])
    _Float16* dstb = &ldsBt[p][0];
    #pragma unroll
    for (int i = 0; i < CH; ++i)
      dstb[(lq * CH + i) * 32 + lk] = breg[i];
    __syncthreads();
    // stage next K-tile while computing this one
    if (kb + 32 < Kpad) {
      const _Float16* brow = Bm + (size_t)(kb + 32 + lk) * Nn + lq * CH;
      #pragma unroll
      for (int i = 0; i < CH; ++i) breg[i] = brow[i];
    }
    // A: lanes 0-15 rows M0..15 K{0-7,16-23}; lanes 16-31 same rows K{8-15,24-31}
    v8h a0 = *(const v8h*)(arow + kb + khalf * 8);
    v8h a1 = *(const v8h*)(arow + kb + 16 + khalf * 8);
    v16h av;
    #pragma unroll
    for (int h = 0; h < 8; ++h) { av[h] = a0[h]; av[h + 8] = a1[h]; }
    __builtin_prefetch(arow + kb + 32, 0, 3);
    const _Float16* lbase = &ldsBt[p][0];
    #pragma unroll
    for (int t = 0; t < NT; ++t) {
      // B: lanes 0-15 K{0-7,16-23}; lanes 16-31 K{8-15,24-31}, col = t*16 + col16
      const _Float16* bl = lbase + (t * 16 + col16) * 32 + khalf * 8;
      v8h b0 = *(const v8h*)(bl);
      v8h b1 = *(const v8h*)(bl + 16);
      v16h bv;
      #pragma unroll
      for (int h = 0; h < 8; ++h) { bv[h] = b0[h]; bv[h + 8] = b1[h]; }
      acc[t] = __builtin_amdgcn_wmma_f32_16x16x32_f16(false, av, false, bv,
                                                      (short)0, acc[t], false, false);
    }
    p ^= 1;
    // ping-pong: iter i+2's store into buffer p is separated from iter i's
    // compute on buffer p by the barrier of iter i+1 -> 1 barrier/K-step.
  }
  #pragma unroll
  for (int t = 0; t < NT; ++t) {
    #pragma unroll
    for (int r = 0; r < 8; ++r) {
      int m = m0 + r + khalf * 8;
      int nn = t * 16 + col16;
      if (m < M) {
        float v = acc[t][r] + (bias ? bias[nn] : 0.f);
        if (act) v = (v > 0.f) ? v : (expf(v) - 1.f);
        C[(size_t)m * Nn + nn] = v;
      }
    }
  }
}

__global__ void k_logsoftmax(const float* logits, long long Brows, int ld, float* out) {
  long long b = (long long)blockIdx.x * blockDim.x + threadIdx.x;
  if (b >= Brows) return;
  float m = -1e30f;
  #pragma unroll
  for (int j = 0; j < 10; ++j) m = fmaxf(m, logits[b * ld + j]);
  float s = 0.f;
  #pragma unroll
  for (int j = 0; j < 10; ++j) s += expf(logits[b * ld + j] - m);
  float lse = m + logf(s);
  #pragma unroll
  for (int j = 0; j < 10; ++j) out[b * 10 + j] = logits[b * ld + j] - lse;
}

// ---------------------------------------------------------------------------
static inline long long llmax3(long long a, long long b, long long c) {
  long long m = a > b ? a : b;
  return m > c ? m : c;
}

extern "C" void kernel_launch(void* const* d_in, const int* in_sizes, int n_in,
                              void* d_out, int out_size, void* d_ws, size_t ws_size,
                              hipStream_t stream) {
  (void)n_in; (void)out_size; (void)ws_size;
  const float* x   = (const float*)d_in[0];
  const float* pos = (const float*)d_in[1];
  const int*   src = (const int*)d_in[2];
  const int*   dst = (const int*)d_in[3];
  const float* W1 = (const float*)d_in[4];
  const float* r1 = (const float*)d_in[5];
  const float* b1 = (const float*)d_in[6];
  const float* W2 = (const float*)d_in[7];
  const float* r2 = (const float*)d_in[8];
  const float* b2 = (const float*)d_in[9];
  const float* W3 = (const float*)d_in[10];
  const float* r3 = (const float*)d_in[11];
  const float* b3 = (const float*)d_in[12];
  const float* fw1 = (const float*)d_in[13];
  const float* fb1 = (const float*)d_in[14];
  const float* fw2 = (const float*)d_in[15];
  const float* fb2 = (const float*)d_in[16];

  long long N  = in_sizes[0];        // nodes (x is [N,1])
  long long Bb = N / 75;             // graphs
  long long E  = in_sizes[2];        // edges
  long long S1 = Bb * 36, S2 = Bb * 25, S3 = Bb * 4;

  // ---- workspace carving ----
  char* wbase = (char*)d_ws;
  size_t off = 0;
  auto carve = [&](size_t bytes) -> char* {
    char* p = wbase + off;
    off = (off + bytes + 255) & ~(size_t)255;
    return p;
  };
  unsigned* scal  = (unsigned*)carve(64);                 // amax0/1/2 (encoded)
  long long accN  = llmax3(N * 25, S1 * 800, S2 * 1600);
  float* acc      = (float*)carve((size_t)accN * 4);
  float* deg      = (float*)carve((size_t)N * 4);
  float* out1     = (float*)carve((size_t)N * 32 * 4);
  float* px1      = (float*)carve((size_t)S1 * 32 * 4);   // enc, decoded in place
  float* ppos1    = (float*)carve((size_t)S1 * 2 * 4);
  float* cnt1     = (float*)carve((size_t)S1 * 4);
  float* sval1    = (float*)carve((size_t)S1 * 4);
  int*   cl1      = (int*)carve((size_t)N * 4);
  unsigned* tab1  = (unsigned*)carve((size_t)S1 * 36 * 4);
  int*   esrc1    = (int*)carve((size_t)E * 4);
  int*   edst1    = (int*)carve((size_t)E * 4);
  float* kf1      = (float*)carve((size_t)E * 4);
  float* out2     = (float*)carve((size_t)S1 * 64 * 4);
  float* px2      = (float*)carve((size_t)S2 * 64 * 4);
  float* ppos2    = (float*)carve((size_t)S2 * 2 * 4);
  float* cnt2     = (float*)carve((size_t)S2 * 4);
  float* sval2    = (float*)carve((size_t)S2 * 4);
  int*   cl2      = (int*)carve((size_t)S1 * 4);
  unsigned* tab2  = (unsigned*)carve((size_t)S2 * 25 * 4);
  int*   esrc2    = (int*)carve((size_t)E * 4);
  int*   edst2    = (int*)carve((size_t)E * 4);
  float* kf2      = (float*)carve((size_t)E * 4);
  float* out3     = (float*)carve((size_t)S2 * 64 * 4);
  float* px3      = (float*)carve((size_t)S3 * 64 * 4);
  float* ppos3    = (float*)carve((size_t)S3 * 2 * 4);
  float* cnt3     = (float*)carve((size_t)S3 * 4);
  int*   cl3      = (int*)carve((size_t)S2 * 4);
  long long afN   = llmax3(N * 32, S1 * 832, S2 * 1664);
  if (Bb * 256 > afN) afN = Bb * 256;
  _Float16* Af    = (_Float16*)carve((size_t)afN * 2);
  _Float16* Bf    = (_Float16*)carve((size_t)1664 * 64 * 2);
  float* hbuf     = (float*)carve((size_t)Bb * 128 * 4);
  float* logits   = (float*)carve((size_t)Bb * 16 * 4);
  float* fb2pad   = (float*)carve(64);

  auto nb = [](long long n) { return dim3((unsigned)((n + TPB - 1) / TPB)); };
  auto gemm = [&](const _Float16* Ag, const _Float16* Bg, const float* bias,
                  float* Cg, long long M, int Nn, int Kpad, int act) {
    dim3 grid((unsigned)((M + 63) / 64));
    if (Nn == 16)
      k_gemm_t<1><<<grid, 128, 0, stream>>>(Ag, Bg, bias, Cg, (int)M, Kpad, act);
    else if (Nn == 32)
      k_gemm_t<2><<<grid, 128, 0, stream>>>(Ag, Bg, bias, Cg, (int)M, Kpad, act);
    else if (Nn == 64)
      k_gemm_t<4><<<grid, 128, 0, stream>>>(Ag, Bg, bias, Cg, (int)M, Kpad, act);
    else  // 128
      k_gemm_t<8><<<grid, 128, 0, stream>>>(Ag, Bg, bias, Cg, (int)M, Kpad, act);
  };

  // ---- stage 0: amax of raw carts ----
  k_fill_u32<<<1, 64, 0, stream>>>(scal, 0u, 16);
  k_amax0<<<nb(E), TPB, 0, stream>>>(pos, src, dst, E, &scal[0]);

  // ---- layer 1: scatter + GEMM [N,32]x[32,32] ----
  k_fill_u32<<<nb(N * 25), TPB, 0, stream>>>((unsigned*)acc, 0u, N * 25);
  k_fill_u32<<<nb(N), TPB, 0, stream>>>((unsigned*)deg, 0u, N);
  k_scatter1<<<nb(E), TPB, 0, stream>>>(x, pos, src, dst, E, &scal[0], acc, deg);
  k_packB<<<nb(32 * 32), TPB, 0, stream>>>(W1, r1, 25, 1, 32, 32, 0, 0, 0, Bf);
  k_packA<<<nb(N * 32), TPB, 0, stream>>>(acc, deg, x, N, 25, 1, 32, Af);
  gemm(Af, Bf, b1, out1, N, 32, 32, 1);

  // ---- pool 1 (size=5, G=6) ----
  k_fill_u32<<<nb(S1 * 32), TPB, 0, stream>>>((unsigned*)px1, 0u, S1 * 32);
  k_fill_u32<<<nb(S1), TPB, 0, stream>>>((unsigned*)cnt1, 0u, S1);
  k_fill_u32<<<nb(S1 * 2), TPB, 0, stream>>>((unsigned*)ppos1, 0u, S1 * 2);
  k_pool_assign<<<nb(N * 32), TPB, 0, stream>>>(out1, pos, nullptr, N, 32, 75,
                                                1.f / 5.f, 6, (unsigned*)px1, cnt1, ppos1, cl1);
  k_pool_final<<<nb(S1 * 32), TPB, 0, stream>>>((unsigned*)px1, cnt1, S1, 32, px1, ppos1, sval1);
  k_fill_u32<<<nb(S1 * 36), TPB, 0, stream>>>(tab1, 0xFFFFFFFFu, S1 * 36);
  k_edge_remap<<<nb(E), TPB, 0, stream>>>(src, dst, nullptr, cl1, E, 36, tab1);
  k_fill_u32<<<nb(S1), TPB, 0, stream>>>((unsigned*)deg, 0u, S1);
  k_edge_keep<<<nb(E), TPB, 0, stream>>>(src, dst, nullptr, cl1, E, 36, tab1, ppos1,
                                         esrc1, edst1, kf1, deg, &scal[1]);

  // ---- layer 2: scatter + GEMM [S1,832]x[832,64] ----
  k_fill_u32<<<nb(S1 * 800), TPB, 0, stream>>>((unsigned*)acc, 0u, S1 * 800);
  k_scatter<<<nb(E * 32), TPB, 0, stream>>>(px1, ppos1, esrc1, edst1, kf1, E, 32, 5,
                                            &scal[1], acc);
  k_packB<<<nb(832 * 64), TPB, 0, stream>>>(W2, r2, 800, 32, 832, 64, 0, 0, 0, Bf);
  k_packA<<<nb(S1 * 832), TPB, 0, stream>>>(acc, deg, px1, S1, 800, 32, 832, Af);
  gemm(Af, Bf, b2, out2, S1, 64, 832, 1);

  // ---- pool 2 (size=7, G=5) ----
  k_fill_u32<<<nb(S2 * 64), TPB, 0, stream>>>((unsigned*)px2, 0u, S2 * 64);
  k_fill_u32<<<nb(S2), TPB, 0, stream>>>((unsigned*)cnt2, 0u, S2);
  k_fill_u32<<<nb(S2 * 2), TPB, 0, stream>>>((unsigned*)ppos2, 0u, S2 * 2);
  k_pool_assign<<<nb(S1 * 64), TPB, 0, stream>>>(out2, ppos1, sval1, S1, 64, 36,
                                                 1.f / 7.f, 5, (unsigned*)px2, cnt2, ppos2, cl2);
  k_pool_final<<<nb(S2 * 64), TPB, 0, stream>>>((unsigned*)px2, cnt2, S2, 64, px2, ppos2, sval2);
  k_fill_u32<<<nb(S2 * 25), TPB, 0, stream>>>(tab2, 0xFFFFFFFFu, S2 * 25);
  k_edge_remap<<<nb(E), TPB, 0, stream>>>(esrc1, edst1, kf1, cl2, E, 25, tab2);
  k_fill_u32<<<nb(S2), TPB, 0, stream>>>((unsigned*)deg, 0u, S2);
  k_edge_keep<<<nb(E), TPB, 0, stream>>>(esrc1, edst1, kf1, cl2, E, 25, tab2, ppos2,
                                         esrc2, edst2, kf2, deg, &scal[2]);

  // ---- layer 3: scatter + GEMM [S2,1664]x[1664,64] ----
  k_fill_u32<<<nb(S2 * 1600), TPB, 0, stream>>>((unsigned*)acc, 0u, S2 * 1600);
  k_scatter<<<nb(E * 64), TPB, 0, stream>>>(px2, ppos2, esrc2, edst2, kf2, E, 64, 6,
                                            &scal[2], acc);
  k_packB<<<nb(1664 * 64), TPB, 0, stream>>>(W3, r3, 1600, 64, 1664, 64, 0, 0, 0, Bf);
  k_packA<<<nb(S2 * 1664), TPB, 0, stream>>>(acc, deg, px2, S2, 1600, 64, 1664, Af);
  gemm(Af, Bf, b3, out3, S2, 64, 1664, 1);

  // ---- pool 3: max_pool_x (size=14, G=2) ----
  k_fill_u32<<<nb(S3 * 64), TPB, 0, stream>>>((unsigned*)px3, 0u, S3 * 64);
  k_fill_u32<<<nb(S3), TPB, 0, stream>>>((unsigned*)cnt3, 0u, S3);
  k_fill_u32<<<nb(S3 * 2), TPB, 0, stream>>>((unsigned*)ppos3, 0u, S3 * 2);
  k_pool_assign<<<nb(S2 * 64), TPB, 0, stream>>>(out3, ppos2, sval2, S2, 64, 25,
                                                 1.f / 14.f, 2, (unsigned*)px3, cnt3, ppos3, cl3);
  k_pool_final<<<nb(S3 * 64), TPB, 0, stream>>>((unsigned*)px3, cnt3, S3, 64, px3, nullptr, nullptr);

  // ---- FC head ----
  k_convert_f16<<<nb(S3 * 64), TPB, 0, stream>>>(px3, Af, S3 * 64);  // [B, 256]
  k_packB<<<nb(256 * 128), TPB, 0, stream>>>(fw1, nullptr, 256, 0, 256, 128, 1, 256, 128, Bf);
  gemm(Af, Bf, fb1, hbuf, Bb, 128, 256, 1);
  k_convert_f16<<<nb(Bb * 128), TPB, 0, stream>>>(hbuf, Af, Bb * 128);
  k_packB<<<nb(128 * 16), TPB, 0, stream>>>(fw2, nullptr, 128, 0, 128, 16, 1, 128, 10, Bf);
  k_padvec<<<1, 16, 0, stream>>>(fb2, 10, 16, fb2pad);
  gemm(Af, Bf, fb2pad, logits, Bb, 16, 128, 0);
  k_logsoftmax<<<nb(Bb), TPB, 0, stream>>>(logits, Bb, 16, (float*)d_out);
}
